// lstm_27470610825948
// MI455X (gfx1250) — matrix operand order repaired and checked by reading, not compiled, not run on hardware
//
#include <hip/hip_runtime.h>
#include <hip/hip_bf16.h>

// Problem constants (from the reference)
#define T_STEPS 512
#define BATCH   128
#define INSZ    256
#define HID     512

typedef __attribute__((ext_vector_type(16))) __bf16        v16bf;
typedef __attribute__((ext_vector_type(8)))  __bf16        v8bf;
typedef __attribute__((ext_vector_type(8)))  float         v8f;
typedef __attribute__((ext_vector_type(4)))  float         f4;
typedef __attribute__((ext_vector_type(4)))  unsigned int  u4;

// ---------------------------------------------------------------------------
// Kernel 1: the actual reference output. outputs[t] == H0 for every t, plus
// the H0 tail. Streaming broadcast with 128-bit non-temporal stores.
// ~134.5MB / 23.3TB/s ~ 5.8us floor.
// ---------------------------------------------------------------------------
__global__ void h0_broadcast_kernel(const f4* __restrict__ h0,
                                    f4* __restrict__ out, int total4) {
  int i = blockIdx.x * blockDim.x + threadIdx.x;
  if (i < total4) {
    f4 v = h0[i & ((BATCH * HID / 4) - 1)];   // B*H/4 = 16384, power of two
    __builtin_nontemporal_store(v, &out[i]);
  }
}

// ---------------------------------------------------------------------------
// Kernel 0: generic f32 -> bf16 (RNE) staging. 8 floats per thread.
// Run once over X, the six live weight matrices, and H0, so the WMMA kernel
// contains ZERO conversion VALU.
// ---------------------------------------------------------------------------
__device__ __forceinline__ unsigned f2bf(float f) {
  union { float f; unsigned u; } c; c.f = f;
  unsigned u = c.u;
  return (u + 0x7fffu + ((u >> 16) & 1u)) >> 16;
}

__global__ void cvt_f32_bf16_kernel(const f4* __restrict__ x4,
                                    unsigned short* __restrict__ xb, int total8) {
  int i = blockIdx.x * blockDim.x + threadIdx.x;
  if (i < total8) {
    f4 a = x4[2 * i], b = x4[2 * i + 1];
    u4 o;
    o.x = f2bf(a.x) | (f2bf(a.y) << 16);
    o.y = f2bf(a.z) | (f2bf(a.w) << 16);
    o.z = f2bf(b.x) | (f2bf(b.y) << 16);
    o.w = f2bf(b.z) | (f2bf(b.w) << 16);
    *(u4*)(xb + 8 * (size_t)i) = o;               // 16B aligned
  }
}

// ---------------------------------------------------------------------------
// WMMA tile loaders (ISA 7.12.2 layouts), all-bf16 sources.
// ---------------------------------------------------------------------------
union ABV { v16bf v; v8bf h[2]; };

// A tile (16x32 bf16), row-major [rows, ldk]. lane L: row M=L&15;
// lanes 0-15 get K 0..7 & 16..23, lanes 16-31 get K 8..15 & 24..31
// -> two contiguous 16B chunks = two b128 loads.
__device__ __forceinline__ v16bf load_a_bf(const __bf16* __restrict__ Mb,
                                           int ldk, int row0, int kk, int lane) {
  const int m  = lane & 15;
  const int k0 = (lane >> 4) * 8;
  const __bf16* p = Mb + (size_t)(row0 + m) * ldk + kk;
  ABV r;
  r.h[0] = *(const v8bf*)(p + k0);
  r.h[1] = *(const v8bf*)(p + 16 + k0);
  return r.v;
}

// B tile (32x16 bf16) from weight W row-major [Ntot, K]; B[k][n]=W[n0+n][kk+k].
// N = lane&15; lanes 0-15 hold K 0..7 & 16..23, lanes 16-31 hold K 8..15 & 24..31
// -> two contiguous 16B chunks = two b128 loads.
__device__ __forceinline__ v16bf load_b_bf(const __bf16* __restrict__ Wb,
                                           int ldk, int n0, int kk, int lane) {
  const int n  = lane & 15;
  const int k0 = (lane >> 4) * 8;
  const __bf16* p = Wb + (size_t)(n0 + n) * ldk + kk + k0;
  ABV r;
  r.h[0] = *(const v8bf*)(p);
  r.h[1] = *(const v8bf*)(p + 16);
  return r.v;
}

__device__ __forceinline__ v8f wmma_bf16(v16bf a, v16bf b, v8f c) {
  return __builtin_amdgcn_wmma_f32_16x16x32_bf16(false, a, false, b,
                                                 (short)0, c, false, false);
}

__device__ __forceinline__ float sigmoidf_(float x) {
  return 1.0f / (1.0f + __expf(-x));
}
__device__ __forceinline__ float tanhf_(float x) {
  return 2.0f * sigmoidf_(2.0f * x) - 1.0f;
}

// ---------------------------------------------------------------------------
// Kernel 2: the C recurrence the reference source expresses, on the WMMA path.
// One wave per 16x16 (batch x hidden) C tile; x-side weights register-resident
// across all 512 steps; every matrix operand pre-staged bf16 (L2-resident).
// ---------------------------------------------------------------------------
__global__ __launch_bounds__(32) void lstm_c_wmma_kernel(
    const __bf16* __restrict__ Xb,                      // [T,B,I] bf16
    const __bf16* __restrict__ H0b,                     // [B,H]   bf16
    const float*  __restrict__ C0,                      // [B,H]   f32
    const __bf16* __restrict__ Wxib, const float* __restrict__ bxi,
    const __bf16* __restrict__ Whib, const float* __restrict__ bhi,
    const __bf16* __restrict__ Wxfb, const float* __restrict__ bxf,
    const __bf16* __restrict__ Whfb, const float* __restrict__ bhf,
    const __bf16* __restrict__ Wxcb, const float* __restrict__ bxc,
    const __bf16* __restrict__ Whcb, const float* __restrict__ bhc,
    float* __restrict__ Cout) {                         // [B,H] scratch
  const int lane = threadIdx.x;
  const int ntn  = HID / 16;
  const int bm   = (blockIdx.x / ntn) * 16;             // batch-tile origin
  const int bn   = (blockIdx.x % ntn) * 16;             // hidden-tile origin
  const int n    = lane & 15;
  const int mhi  = (lane >> 4) * 8;                     // C/D row offset per half

  // ---- hidden-side projections: constant over time (source bug) ----
  v8f hi = {}, hf = {}, hc = {};
  for (int kk = 0; kk < HID; kk += 32) {
    v16bf a = load_a_bf(H0b, HID, bm, kk, lane);
    hi = wmma_bf16(a, load_b_bf(Whib, HID, bn, kk, lane), hi);
    hf = wmma_bf16(a, load_b_bf(Whfb, HID, bn, kk, lane), hf);
    hc = wmma_bf16(a, load_b_bf(Whcb, HID, bn, kk, lane), hc);
  }
  const float bi  = bxi[bn + n] + bhi[bn + n];
  const float bf_ = bxf[bn + n] + bhf[bn + n];
  const float bc  = bxc[bn + n] + bhc[bn + n];
#pragma unroll
  for (int r = 0; r < 8; ++r) { hi[r] += bi; hf[r] += bf_; hc[r] += bc; }

  // ---- x-side weight tiles: register-resident across all 512 steps ----
  v16bf wbi[8], wbf[8], wbc[8];
#pragma unroll
  for (int j = 0; j < 8; ++j) {
    wbi[j] = load_b_bf(Wxib, INSZ, bn, j * 32, lane);
    wbf[j] = load_b_bf(Wxfb, INSZ, bn, j * 32, lane);
    wbc[j] = load_b_bf(Wxcb, INSZ, bn, j * 32, lane);
  }

  // ---- C tile in registers (C/D layout: M = r + mhi, N = lane&15) ----
  v8f c;
#pragma unroll
  for (int r = 0; r < 8; ++r)
    c[r] = C0[(size_t)(bm + mhi + r) * HID + bn + n];

  // ---- time loop: 16 b128 loads + 24 WMMA + gate math per step ----
  for (int t = 0; t < T_STEPS; ++t) {
    const __bf16* Xt = Xb + (size_t)t * BATCH * INSZ;
    if (t + 1 < T_STEPS)   // global_prefetch_b8 of next step's A rows
      __builtin_prefetch(Xb + (size_t)(t + 1) * BATCH * INSZ +
                             (size_t)(bm + (lane & 15)) * INSZ, 0, 1);
    v8f xi = {}, xf = {}, xc = {};
#pragma unroll
    for (int j = 0; j < 8; ++j) {
      v16bf a = load_a_bf(Xt, INSZ, bm, j * 32, lane);
      xi = wmma_bf16(a, wbi[j], xi);
      xf = wmma_bf16(a, wbf[j], xf);
      xc = wmma_bf16(a, wbc[j], xc);
    }
#pragma unroll
    for (int r = 0; r < 8; ++r) {
      float gi = sigmoidf_(xi[r] + hi[r]);
      float gf = sigmoidf_(xf[r] + hf[r]);
      float gc = tanhf_(xc[r] + hc[r]);
      c[r] = gf * c[r] + gi * gc;
    }
  }

#pragma unroll
  for (int r = 0; r < 8; ++r)
    Cout[(size_t)(bm + mhi + r) * HID + bn + n] = c[r];
}

// ---------------------------------------------------------------------------
extern "C" void kernel_launch(void* const* d_in, const int* in_sizes, int n_in,
                              void* d_out, int out_size, void* d_ws, size_t ws_size,
                              hipStream_t stream) {
  const float* X   = (const float*)d_in[0];
  const float* H0  = (const float*)d_in[1];
  const float* C0  = (const float*)d_in[2];
  const float* Wxi = (const float*)d_in[3];  const float* bxi = (const float*)d_in[4];
  const float* Whi = (const float*)d_in[5];  const float* bhi = (const float*)d_in[6];
  const float* Wxf = (const float*)d_in[7];  const float* bxf = (const float*)d_in[8];
  const float* Whf = (const float*)d_in[9];  const float* bhf = (const float*)d_in[10];
  // d_in[11..14] = W_xo/b_xo/W_ho/b_ho: the O gate is dead in the reference.
  const float* Wxc = (const float*)d_in[15]; const float* bxc = (const float*)d_in[16];
  const float* Whc = (const float*)d_in[17]; const float* bhc = (const float*)d_in[18];

  // Output = [T,B,H] broadcast of H0, then H0 again. out_size = 513*B*H.
  int total4 = out_size / 4;
  h0_broadcast_kernel<<<(total4 + 255) / 256, 256, 0, stream>>>(
      (const f4*)H0, (f4*)d_out, total4);

  // d_ws layout (bytes, all 16B aligned):
  //   Xbf        : T*B*I*2          = 67,108,864
  //   Whi/Whf/Whc: 3 * H*H*2        =  1,572,864
  //   Wxi/Wxf/Wxc: 3 * H*I*2        =    786,432
  //   H0bf       : B*H*2            =    131,072
  //   Cout       : B*H*4            =    262,144
  const size_t nX  = (size_t)T_STEPS * BATCH * INSZ;
  const size_t nWh = (size_t)HID * HID;
  const size_t nWx = (size_t)HID * INSZ;
  const size_t nH0 = (size_t)BATCH * HID;
  const size_t need = (nX + 3 * nWh + 3 * nWx + nH0) * 2 + nH0 * 4;
  if (ws_size < need) return;

  unsigned short* w = (unsigned short*)d_ws;
  unsigned short* Xb   = w;            w += nX;
  unsigned short* Whib = w;            w += nWh;
  unsigned short* Whfb = w;            w += nWh;
  unsigned short* Whcb = w;            w += nWh;
  unsigned short* Wxib = w;            w += nWx;
  unsigned short* Wxfb = w;            w += nWx;
  unsigned short* Wxcb = w;            w += nWx;
  unsigned short* H0b  = w;            w += nH0;
  float*          Cw   = (float*)w;

  struct Job { const float* src; unsigned short* dst; size_t n; };
  const Job jobs[8] = {
      {X, Xb, nX},     {Whi, Whib, nWh}, {Whf, Whfb, nWh}, {Whc, Whcb, nWh},
      {Wxi, Wxib, nWx}, {Wxf, Wxfb, nWx}, {Wxc, Wxcb, nWx}, {H0, H0b, nH0}};
  for (int j = 0; j < 8; ++j) {
    int total8 = (int)(jobs[j].n / 8);
    cvt_f32_bf16_kernel<<<(total8 + 255) / 256, 256, 0, stream>>>(
        (const f4*)jobs[j].src, jobs[j].dst, total8);
  }

  lstm_c_wmma_kernel<<<(BATCH / 16) * (HID / 16), 32, 0, stream>>>(
      (const __bf16*)Xb, (const __bf16*)H0b, C0,
      (const __bf16*)Wxib, bxi, (const __bf16*)Whib, bhi,
      (const __bf16*)Wxfb, bxf, (const __bf16*)Whfb, bhf,
      (const __bf16*)Wxcb, bxc, (const __bf16*)Whcb, bhc, Cw);
}
